// SimCLRLoss_17248588661028
// MI455X (gfx1250) — matrix operand order repaired and checked by reading, not compiled, not run on hardware
//
#include <hip/hip_runtime.h>
#include <hip/hip_fp16.h>

typedef __attribute__((ext_vector_type(16))) _Float16 v16h;
typedef __attribute__((ext_vector_type(8)))  _Float16 v8h;
typedef __attribute__((ext_vector_type(4)))  _Float16 v4h;
typedef __attribute__((ext_vector_type(8)))  float    v8f;
typedef __attribute__((ext_vector_type(4)))  int      v4i;

#define B_ROWS 4096
#define DDIM   128
#define N2     8192      // 2B
#define INV_T  10.0f     // 1 / 0.1
#define EPS_N  1e-12f
#define EXP2_SCALE 14.426950408889634f   // INV_T * log2(e)

#define NTILES        64          // column tiles per block (one chunk)
#define LDS_ROW_HALFS 136         // 128 + 8 pad -> 68-dword row stride, conflict-free

// CDNA5 async global->LDS path if the toolchain exposes the builtins.
#if defined(__has_builtin)
#  if __has_builtin(__builtin_amdgcn_global_load_async_to_lds_b128) && \
      __has_builtin(__builtin_amdgcn_s_wait_asynccnt)
#    define USE_ASYNC_LDS 1
#  endif
#endif
#ifndef USE_ASYNC_LDS
#  define USE_ASYNC_LDS 0
#endif

// exp(x / T) with a single VALU mul feeding v_exp_f32
__device__ __forceinline__ float exp_invT(float x) {
#if defined(__has_builtin) && __has_builtin(__builtin_amdgcn_exp2f)
    return __builtin_amdgcn_exp2f(x * EXP2_SCALE);
#else
    return __expf(x * INV_T);
#endif
}

// ---------------------------------------------------------------------------
// Kernel 1: L2-normalize rows of x_i / x_j, pack to f16 Z[2B,128].
//   pos[i]  = <zi_hat, zj_hat>           (exact f32 positive)
//   diag[r] = sum of squares of the f16-rounded row (== sim_rr of the f16
//             GEMM up to summation order; f16*f16 is exact in f32)
// One wave per row pair; each lane owns 4 consecutive floats.
// ---------------------------------------------------------------------------
__global__ void __launch_bounds__(256)
simclr_normalize(const float* __restrict__ xi, const float* __restrict__ xj,
                 _Float16* __restrict__ Z, float* __restrict__ pos,
                 float* __restrict__ diag) {
    const int wave = (blockIdx.x * blockDim.x + threadIdx.x) >> 5;   // row index i
    const int lane = threadIdx.x & 31;
    if (wave >= B_ROWS) return;

    const float4 a = ((const float4*)(xi + (size_t)wave * DDIM))[lane];
    const float4 b = ((const float4*)(xj + (size_t)wave * DDIM))[lane];

    float sa = a.x*a.x + a.y*a.y + a.z*a.z + a.w*a.w;
    float sb = b.x*b.x + b.y*b.y + b.z*b.z + b.w*b.w;
    #pragma unroll
    for (int m = 1; m < 32; m <<= 1) {
        sa += __shfl_xor(sa, m, 32);
        sb += __shfl_xor(sb, m, 32);
    }
    const float ia = 1.0f / fmaxf(sqrtf(sa), EPS_N);
    const float ib = 1.0f / fmaxf(sqrtf(sb), EPS_N);

    const float nax = a.x*ia, nay = a.y*ia, naz = a.z*ia, naw = a.w*ia;
    const float nbx = b.x*ib, nby = b.y*ib, nbz = b.z*ib, nbw = b.w*ib;

    float dp = nax*nbx + nay*nby + naz*nbz + naw*nbw;

    v4h ha, hb;
    ha[0] = (_Float16)nax; ha[1] = (_Float16)nay; ha[2] = (_Float16)naz; ha[3] = (_Float16)naw;
    hb[0] = (_Float16)nbx; hb[1] = (_Float16)nby; hb[2] = (_Float16)nbz; hb[3] = (_Float16)nbw;

    // f16-rounded self dot products (what the WMMA diagonal sees)
    float da = 0.0f, db = 0.0f;
    #pragma unroll
    for (int k = 0; k < 4; ++k) {
        const float fa = (float)ha[k], fb = (float)hb[k];
        da += fa * fa;
        db += fb * fb;
    }
    #pragma unroll
    for (int m = 1; m < 32; m <<= 1) {
        dp += __shfl_xor(dp, m, 32);
        da += __shfl_xor(da, m, 32);
        db += __shfl_xor(db, m, 32);
    }

    *(v4h*)(Z + (size_t)wave          * DDIM + lane * 4) = ha;
    *(v4h*)(Z + (size_t)(B_ROWS+wave) * DDIM + lane * 4) = hb;
    if (lane == 0) {
        pos[wave]           = dp;
        diag[wave]          = da;
        diag[B_ROWS + wave] = db;
    }
}

// ---------------------------------------------------------------------------
// Cooperative stage of one 16x128(f16) column tile (4 KB) into LDS.
// 256 threads, one b128 chunk each. Async (ASYNCcnt) when available.
// ---------------------------------------------------------------------------
__device__ __forceinline__ void stage_tile(const _Float16* __restrict__ Z, int n0,
                                           _Float16* dst) {
    const int tid = threadIdx.x;
    const int row = tid >> 4;        // 0..15
    const int cir = tid & 15;        // 16-byte chunk within the 256B row
    const _Float16* src = Z + (size_t)(n0 + row) * DDIM + cir * 8;
    _Float16* d = dst + row * LDS_ROW_HALFS + cir * 8;
#if USE_ASYNC_LDS
    __builtin_amdgcn_global_load_async_to_lds_b128(
        (__attribute__((address_space(1))) v4i*)src,
        (__attribute__((address_space(3))) v4i*)d, 0, 0);
#else
    *(v8h*)d = *(const v8h*)src;
#endif
}

__device__ __forceinline__ void stage_fence() {
#if USE_ASYNC_LDS
    __builtin_amdgcn_s_wait_asynccnt(0);
#endif
    __syncthreads();
}

// ---------------------------------------------------------------------------
// Kernel 2: fused  C = Z * Z^T  (f16 WMMA, f32 accum)  +  rowsum of
//           exp(C/T) — full rows, self term subtracted later via diag[].
// Block (8 waves) = 128-row block x one chunk of 64 column tiles.
// Column tiles staged once per block in double-buffered LDS (async DMA),
// shared by all 8 waves. Hot loop is branch-free:
//   8x ds_load_b128 (one clause/wait) -> 4 chained WMMAs -> 8 exp -> pk_adds
// ---------------------------------------------------------------------------
__global__ void __launch_bounds__(256)
simclr_sim_reduce(const _Float16* __restrict__ Z, float* __restrict__ rowsum) {
    __shared__ _Float16 lbuf[2][16 * LDS_ROW_HALFS];

    const int lane   = threadIdx.x & 31;
    const int wave   = threadIdx.x >> 5;
    const int rowblk = blockIdx.x >> 3;       // 0..63  (128 rows each)
    const int cchunk = blockIdx.x & 7;        // 0..7   (64 column tiles each)
    const int m0     = rowblk * 128 + wave * 16;
    const int half   = lane >> 4;
    const int l15    = lane & 15;

    // Hoist A fragments for the whole K=128 (4 steps of 32)
    v16h afrag[4];
    const _Float16* arow = Z + (size_t)(m0 + l15) * DDIM;
    #pragma unroll
    for (int kk = 0; kk < 4; ++kk) {
        const int k0 = kk * 32;
        union { v16h v; v8h h[2]; } u;
        u.h[0] = *(const v8h*)(arow + k0 + 8 * half);
        u.h[1] = *(const v8h*)(arow + k0 + 16 + 8 * half);
        afrag[kk] = u.v;
    }

    float acc[8];
    #pragma unroll
    for (int g = 0; g < 8; ++g) acc[g] = 0.0f;

    // Prologue: stage tile 0
    stage_tile(Z, (cchunk * NTILES + 0) * 16, lbuf[0]);
    stage_fence();

    for (int t = 0; t < NTILES; ++t) {
        const int cur = t & 1;

        // Prefetch next tile into the other buffer (fully consumed at t-1)
        if (t + 1 < NTILES)
            stage_tile(Z, (cchunk * NTILES + t + 1) * 16, lbuf[cur ^ 1]);

        // All 4 B fragments up front: one ds clause, one wait
        const _Float16* brow = &lbuf[cur][l15 * LDS_ROW_HALFS];
        v16h bfrag[4];
        #pragma unroll
        for (int kk = 0; kk < 4; ++kk) {
            union { v16h v; v8h h[2]; } u;
            u.h[0] = *(const v8h*)(brow + kk * 32 + 8 * half);
            u.h[1] = *(const v8h*)(brow + kk * 32 + 16 + 8 * half);
            bfrag[kk] = u.v;
        }

        v8f c = {};
        #pragma unroll
        for (int kk = 0; kk < 4; ++kk)
            c = __builtin_amdgcn_wmma_f32_16x16x32_f16(
                    false, afrag[kk], false, bfrag[kk],
                    (short)0, c, false, false);

        #pragma unroll
        for (int g = 0; g < 8; ++g)
            acc[g] += exp_invT(c[g]);

        stage_fence();   // staging of t+1 complete, all waves done reading cur
    }

    // Reduce across the 16 lanes of each half (bit 4 untouched by masks 1..8)
    #pragma unroll
    for (int g = 0; g < 8; ++g) {
        float v = acc[g];
        v += __shfl_xor(v, 1, 32);
        v += __shfl_xor(v, 2, 32);
        v += __shfl_xor(v, 4, 32);
        v += __shfl_xor(v, 8, 32);
        acc[g] = v;
    }
    if (l15 == 0) {
        #pragma unroll
        for (int g = 0; g < 8; ++g)
            atomicAdd(&rowsum[m0 + g + 8 * half], acc[g]);
    }
}

// ---------------------------------------------------------------------------
// Kernel 3: denom[r] = rowsum[r] - exp(diag[r]/T)   (remove self term)
//           loss = ( sum_r log(denom[r]) - 2 * sum_i pos[i]/T ) / (2B)
// ---------------------------------------------------------------------------
__global__ void __launch_bounds__(256)
simclr_final(const float* __restrict__ rowsum, const float* __restrict__ diag,
             const float* __restrict__ pos, float* __restrict__ out) {
    __shared__ float sm[256];
    float s = 0.0f;
    for (int r = threadIdx.x; r < N2; r += 256)
        s += __logf(rowsum[r] - exp_invT(diag[r]));
    for (int i = threadIdx.x; i < B_ROWS; i += 256)
        s -= 2.0f * INV_T * pos[i];
    sm[threadIdx.x] = s;
    __syncthreads();
    #pragma unroll
    for (int o = 128; o > 0; o >>= 1) {
        if (threadIdx.x < o) sm[threadIdx.x] += sm[threadIdx.x + o];
        __syncthreads();
    }
    if (threadIdx.x == 0) out[0] = sm[0] / (float)N2;
}

// ---------------------------------------------------------------------------
extern "C" void kernel_launch(void* const* d_in, const int* in_sizes, int n_in,
                              void* d_out, int out_size, void* d_ws, size_t ws_size,
                              hipStream_t stream) {
    const float* xi = (const float*)d_in[0];
    const float* xj = (const float*)d_in[1];
    float* out = (float*)d_out;

    char* ws = (char*)d_ws;
    _Float16* Z    = (_Float16*)ws;                                        // 2 MiB
    float*    pos  = (float*)(ws + (size_t)N2 * DDIM * sizeof(_Float16)); // 16 KiB
    float*    rsum = pos + B_ROWS;                                         // 32 KiB
    float*    diag = rsum + N2;                                            // 32 KiB

    (void)hipMemsetAsync(rsum, 0, (size_t)N2 * sizeof(float), stream);

    simclr_normalize<<<B_ROWS / 8, 256, 0, stream>>>(xi, xj, Z, pos, diag);
    simclr_sim_reduce<<<512, 256, 0, stream>>>(Z, rsum);
    simclr_final<<<1, 256, 0, stream>>>(rsum, diag, pos, out);
}
